// MultiHeadAttention_17214228922650
// MI455X (gfx1250) — matrix operand order repaired and checked by reading, not compiled, not run on hardware
//
#include <hip/hip_runtime.h>
#include <hip/hip_bf16.h>

// MHA forward on gfx1250: all matmuls via v_wmma_f32_16x16x32_f16 (f32 accum),
// flash-attention (online softmax) so the [S,S] score matrix never hits HBM,
// K/V tiles staged into LDS by the Tensor Data Mover (double-buffered,
// TENSORcnt-tracked) so the 4 waves of a block share one copy.

typedef __attribute__((ext_vector_type(16))) _Float16 v16h;
typedef __attribute__((ext_vector_type(8)))  _Float16 v8h;
typedef __attribute__((ext_vector_type(4)))  _Float16 v4h;
typedef __attribute__((ext_vector_type(8)))  float    v8f;
typedef __attribute__((ext_vector_type(4)))  unsigned v4u;
typedef __attribute__((ext_vector_type(8)))  int      v8i_;
typedef __attribute__((ext_vector_type(4)))  int      v4i_;

#define B_  4
#define S_  2048
#define H_  1024
#define NH_ 16
#define HD_ 64

#if defined(__has_builtin)
#if __has_builtin(__builtin_amdgcn_tensor_load_to_lds) && \
    __has_builtin(__builtin_amdgcn_s_wait_tensorcnt)
#define USE_TDM 1
#endif
#endif
#ifndef USE_TDM
#define USE_TDM 0
#endif

__device__ __forceinline__ v8f wmma16(v16h a, v16h b, v8f c) {
  // D = A(16x32 f16) * B(32x16 f16) + C(16x16 f32)
  return __builtin_amdgcn_wmma_f32_16x16x32_f16(false, a, false, b, (short)0, c,
                                                false, false);
}

// A fragment 16x32 (MxK), row-major source with leading dim ld (halfs).
// lanes 0-15 row M=lane hold K=0..7 / K=16..23; lanes 16-31 hold K=8..15 / 24..31.
__device__ __forceinline__ v16h load_a(const _Float16* base, int ld, int lane) {
  int r = lane & 15;
  const _Float16* p = base + (size_t)r * ld + ((lane < 16) ? 0 : 8);
  v8h lo = *(const v8h*)p;
  v8h hi = *(const v8h*)(p + 16);
  v16h out;
#pragma unroll
  for (int i = 0; i < 8; ++i) { out[i] = lo[i]; out[8 + i] = hi[i]; }
  return out;
}

// B fragment 32x16 (KxN): lane n<16 holds column n K=0..7 / 16..23; lane n+16 K=8..15 / 24..31.
// p = pointer to (k=0, column = lane&15) contiguous k-run.
__device__ __forceinline__ v16h load_b(const _Float16* p, int lane) {
  p += (lane < 16) ? 0 : 8;
  v8h lo = *(const v8h*)p;
  v8h hi = *(const v8h*)(p + 16);
  v16h out;
#pragma unroll
  for (int i = 0; i < 8; ++i) { out[i] = lo[i]; out[8 + i] = hi[i]; }
  return out;
}

// LDS byte offset of a __shared__ object (generic -> AS(3) -> int)
__device__ __forceinline__ unsigned lds_off(const void* p) {
  typedef __attribute__((address_space(3))) const void* as3p;
  return (unsigned)(unsigned long long)(as3p)p;
}

// TDM D# group0: count=1, lds_addr, 57-bit global byte address, type=2
__device__ __forceinline__ v4u tdm_g0(unsigned ldsbyte, const void* gptr) {
  unsigned long long ga = (unsigned long long)gptr;
  v4u g0;
  g0[0] = 1u;  // count=1, user mode, no gather
  g0[1] = ldsbyte;
  g0[2] = (unsigned)ga;
  g0[3] = ((unsigned)(ga >> 32) & 0x01FFFFFFu) | 0x80000000u;  // type=2
  return g0;
}

// ---------------- fp32 -> fp16 conversion (vectorized x4) ----------------
__global__ void cvt16(const float* __restrict__ in, _Float16* __restrict__ out,
                      int n) {
  int i = (blockIdx.x * blockDim.x + threadIdx.x) * 4;
  if (i + 3 < n) {
    float4 f = *(const float4*)(in + i);
    v4h h;
    h[0] = (_Float16)f.x; h[1] = (_Float16)f.y;
    h[2] = (_Float16)f.z; h[3] = (_Float16)f.w;
    *(v4h*)(out + i) = h;
  }
}

// ---------------- GEMM: Y[M,N] = X[M,K] @ W[N,K]^T -----------------------
// Each wave: 32x64 strip (2 A-frags reuse every B-frag -> 8 wmma / k-step).
// mode 0: f16 out, head-split  [B,NH,S,HD]   (Q, K projections)
// mode 1: f16 out, transposed  [B,NH,HD,S]   (V projection)
// mode 2: f32 out + bias       [M,N]         (output projection)
__global__ __launch_bounds__(128) void gemm_xwt(
    const _Float16* __restrict__ X, const _Float16* __restrict__ W,
    _Float16* __restrict__ dstH, float* __restrict__ dstF,
    const float* __restrict__ bias, int mode) {
  const int K = H_;
  int lane = threadIdx.x & 31;
  int wave = threadIdx.x >> 5;
  int tileM = blockIdx.x * 128 + wave * 32;
  int tileN = blockIdx.y * 64;
  int col = lane & 15;

  const _Float16* Arow0 = X + (size_t)tileM * K;
  const _Float16* Arow1 = Arow0 + (size_t)16 * K;
  const _Float16* Bcol = W + (size_t)(tileN + col) * K;  // B col n == W row n

  v8f acc[2][4];
#pragma unroll
  for (int p = 0; p < 2; ++p)
#pragma unroll
    for (int nb = 0; nb < 4; ++nb)
#pragma unroll
      for (int i = 0; i < 8; ++i) acc[p][nb][i] = 0.0f;

  for (int k0 = 0; k0 < K; k0 += 32) {
    v16h a0 = load_a(Arow0 + k0, K, lane);
    v16h a1 = load_a(Arow1 + k0, K, lane);
#pragma unroll
    for (int nb = 0; nb < 4; ++nb) {
      v16h b = load_b(Bcol + (size_t)nb * 16 * K + k0, lane);
      acc[0][nb] = wmma16(a0, b, acc[0][nb]);
      acc[1][nb] = wmma16(a1, b, acc[1][nb]);
    }
  }

  // C layout: VGPR i -> row M=i (lanes 0-15) / M=8+i (lanes 16-31), col=lane&15
  int rb = (lane < 16) ? 0 : 8;
#pragma unroll
  for (int p = 0; p < 2; ++p) {
#pragma unroll
    for (int nb = 0; nb < 4; ++nb) {
#pragma unroll
      for (int i = 0; i < 8; ++i) {
        int m = tileM + p * 16 + rb + i;
        int n = tileN + nb * 16 + col;
        float v = acc[p][nb][i];
        if (mode == 2) {
          dstF[(size_t)m * H_ + n] = v + bias[n];
        } else {
          int b = m >> 11, s = m & (S_ - 1);
          int h = n >> 6,  d = n & (HD_ - 1);
          size_t idx = (mode == 0)
              ? ((((size_t)b * NH_ + h) * S_ + s) * HD_ + d)
              : ((((size_t)b * NH_ + h) * HD_ + d) * S_ + s);
          dstH[idx] = (_Float16)v;
        }
      }
    }
  }
}

// ---------------- Flash attention over one (b,h) / 64-query tile ----------
__global__ __launch_bounds__(128) void attn_fa(
    const _Float16* __restrict__ Qh, const _Float16* __restrict__ Kh,
    const _Float16* __restrict__ Vt, const int* __restrict__ mask,
    _Float16* __restrict__ ctx) {
  int lane = threadIdx.x & 31;
  int wave = threadIdx.x >> 5;
  int tid  = threadIdx.x;
  int bh = blockIdx.y;
  int b = bh >> 4, h = bh & 15;
  int q0 = blockIdx.x * 64 + wave * 16;  // 16 query rows per wave
  int col = lane & 15;
  int rb = (lane < 16) ? 0 : 8;

  const _Float16* Qb = Qh + ((size_t)bh * S_ + q0) * HD_;
  const _Float16* Kb = Kh + (size_t)bh * S_ * HD_;
  const _Float16* Vb = Vt + (size_t)bh * HD_ * S_;
  const int* mb = mask + (size_t)b * S_ * S_ + (size_t)(q0 + rb) * S_;

  __shared__ __align__(16) _Float16 kbuf[2][32 * 64];  // [key][dim]
  __shared__ __align__(16) _Float16 vbuf[2][64 * 32];  // [dim][key]
  __shared__ __align__(16) _Float16 plds[4][16][32];   // C->A prob re-layout

#if USE_TDM
  // D# group1 for K tile: 2D tensor [S_ rows x HD_], tile 32x64, 2B elements
  v8i_ g1k; g1k[0] = 1 << 16; g1k[1] = HD_ << 16; g1k[2] = S_ << 16;
  g1k[3] = HD_ << 16; g1k[4] = 32; g1k[5] = HD_; g1k[6] = 0; g1k[7] = 0;
  // D# group1 for V tile: 2D tensor [HD_ rows x S_], tile 64x32
  v8i_ g1v; g1v[0] = 1 << 16; g1v[1] = S_ << 16; g1v[2] = HD_ << 16;
  g1v[3] = 32 << 16; g1v[4] = HD_; g1v[5] = S_; g1v[6] = 0; g1v[7] = 0;
  v4i_ z4; z4[0] = 0; z4[1] = 0; z4[2] = 0; z4[3] = 0;
  v8i_ z8; z8[0] = 0; z8[1] = 0; z8[2] = 0; z8[3] = 0;
  z8[4] = 0; z8[5] = 0; z8[6] = 0; z8[7] = 0;

  // prologue: wave 0 DMAs tile 0 into buffer 0 (2 TDM ops, TENSORcnt += 2)
  if (wave == 0) {
    __builtin_amdgcn_tensor_load_to_lds(tdm_g0(lds_off(&kbuf[0][0]), Kb),
                                        g1k, z4, z4, z8, 0);
    __builtin_amdgcn_tensor_load_to_lds(tdm_g0(lds_off(&vbuf[0][0]), Vb),
                                        g1v, z4, z4, z8, 0);
  }
#endif

  // Q fragments for this wave's 16 rows, pre-scaled by 1/sqrt(HD) = 0.125
  v16h qa0 = load_a(Qb, HD_, lane);        // dims 0..31
  v16h qa1 = load_a(Qb + 32, HD_, lane);   // dims 32..63
#pragma unroll
  for (int i = 0; i < 16; ++i) {
    qa0[i] *= (_Float16)0.125f;
    qa1[i] *= (_Float16)0.125f;
  }

  v8f o[4];
  float mrow[8], lrow[8];
#pragma unroll
  for (int nb = 0; nb < 4; ++nb)
#pragma unroll
    for (int i = 0; i < 8; ++i) o[nb][i] = 0.0f;
#pragma unroll
  for (int i = 0; i < 8; ++i) { mrow[i] = -1e30f; lrow[i] = 0.0f; }

  for (int k0 = 0, t = 0; k0 < S_; k0 += 32, ++t) {
    int cur = t & 1;
#if USE_TDM
    if (wave == 0) {
      if (k0 + 32 < S_) {  // prefetch next tile into other buffer
        __builtin_amdgcn_tensor_load_to_lds(
            tdm_g0(lds_off(&kbuf[cur ^ 1][0]), Kb + (size_t)(k0 + 32) * HD_),
            g1k, z4, z4, z8, 0);
        __builtin_amdgcn_tensor_load_to_lds(
            tdm_g0(lds_off(&vbuf[cur ^ 1][0]), Vb + (k0 + 32)),
            g1v, z4, z4, z8, 0);
        __builtin_amdgcn_s_wait_tensorcnt((short)2);  // current tile landed
      } else {
        __builtin_amdgcn_s_wait_tensorcnt((short)0);
      }
    }
#else
    // cooperative synchronous staging fallback
    for (int c = tid; c < 256; c += 128) {
      int key = c >> 3, of = (c & 7) * 8;
      *(v8h*)(&kbuf[cur][key * 64 + of]) =
          *(const v8h*)(Kb + (size_t)(k0 + key) * HD_ + of);
    }
    for (int c = tid; c < 256; c += 128) {
      int d = c >> 2, of = (c & 3) * 8;
      *(v8h*)(&vbuf[cur][d * 32 + of]) =
          *(const v8h*)(Vb + (size_t)d * S_ + k0 + of);
    }
#endif
    __syncthreads();  // publish tile to all waves

    // ---- scores: S = Q * K^T over two 16-key tiles (from LDS) ----
    const _Float16* kc0 = &kbuf[cur][(size_t)col * HD_];
    const _Float16* kc1 = &kbuf[cur][(size_t)(16 + col) * HD_];
    v16h b00 = load_b(kc0, lane);       // keys 0..15,  dims 0..31
    v16h b01 = load_b(kc0 + 32, lane);  // dims 32..63
    v16h b10 = load_b(kc1, lane);       // keys 16..31
    v16h b11 = load_b(kc1 + 32, lane);
    v8f s0, s1;
#pragma unroll
    for (int i = 0; i < 8; ++i) { s0[i] = 0.0f; s1[i] = 0.0f; }
    s0 = wmma16(qa0, b00, s0); s0 = wmma16(qa1, b01, s0);
    s1 = wmma16(qa0, b10, s1); s1 = wmma16(qa1, b11, s1);

    // ---- mask (mask==0 -> -1e9) ----
#pragma unroll
    for (int i = 0; i < 8; ++i) {
      if (mb[(size_t)i * S_ + k0 + col] == 0)      s0[i] = -1e9f;
      if (mb[(size_t)i * S_ + k0 + 16 + col] == 0) s1[i] = -1e9f;
    }

    // ---- online softmax (row stats across the 16 lanes of each half) ----
    v8f tm;
#pragma unroll
    for (int i = 0; i < 8; ++i) tm[i] = fmaxf(s0[i], s1[i]);
#pragma unroll
    for (int off = 8; off >= 1; off >>= 1)
#pragma unroll
      for (int i = 0; i < 8; ++i) tm[i] = fmaxf(tm[i], __shfl_xor(tm[i], off));

    float pr0[8], pr1[8], corr[8];
    v8f ts;
#pragma unroll
    for (int i = 0; i < 8; ++i) {
      float mn = fmaxf(mrow[i], tm[i]);
      corr[i] = __expf(mrow[i] - mn);
      mrow[i] = mn;
      pr0[i] = __expf(s0[i] - mn);
      pr1[i] = __expf(s1[i] - mn);
      ts[i] = pr0[i] + pr1[i];
    }
#pragma unroll
    for (int off = 8; off >= 1; off >>= 1)
#pragma unroll
      for (int i = 0; i < 8; ++i) ts[i] += __shfl_xor(ts[i], off);
#pragma unroll
    for (int i = 0; i < 8; ++i) lrow[i] = lrow[i] * corr[i] + ts[i];
#pragma unroll
    for (int nb = 0; nb < 4; ++nb)
#pragma unroll
      for (int i = 0; i < 8; ++i) o[nb][i] *= corr[i];

    // ---- C-layout probs -> LDS -> A-layout fragment (wave-local) ----
#pragma unroll
    for (int i = 0; i < 8; ++i) {
      plds[wave][rb + i][col]      = (_Float16)pr0[i];
      plds[wave][rb + i][col + 16] = (_Float16)pr1[i];
    }
    asm volatile("s_wait_dscnt 0x0" ::: "memory");  // wave-local LDS RAW
    v16h pa = load_a(&plds[wave][0][0], 32, lane);

    // ---- O += P * V  (B-frag columns = vbuf rows, contiguous over keys) ----
#pragma unroll
    for (int nb = 0; nb < 4; ++nb) {
      v16h vb = load_b(&vbuf[cur][(size_t)(nb * 16 + col) * 32], lane);
      o[nb] = wmma16(pa, vb, o[nb]);
    }

    __syncthreads();  // all waves done reading before buffer reuse
  }

  // ---- normalize and store context as f16 [B,S,H] ----
#pragma unroll
  for (int i = 0; i < 8; ++i) lrow[i] = 1.0f / lrow[i];
  _Float16* cb = ctx + ((size_t)b * S_ + (q0 + rb)) * H_ + h * HD_;
#pragma unroll
  for (int nb = 0; nb < 4; ++nb)
#pragma unroll
    for (int i = 0; i < 8; ++i)
      cb[(size_t)i * H_ + nb * 16 + col] = (_Float16)(o[nb][i] * lrow[i]);
}

// --------------------------------------------------------------------------
extern "C" void kernel_launch(void* const* d_in, const int* in_sizes, int n_in,
                              void* d_out, int out_size, void* d_ws,
                              size_t ws_size, hipStream_t stream) {
  const float* q    = (const float*)d_in[0];
  const float* k    = (const float*)d_in[1];
  const float* v    = (const float*)d_in[2];
  const int*   mask = (const int*)d_in[3];
  const float* Wq   = (const float*)d_in[4];
  const float* Wk   = (const float*)d_in[5];
  const float* Wv   = (const float*)d_in[6];
  const float* Wo   = (const float*)d_in[7];
  const float* bo   = (const float*)d_in[8];
  float* out = (float*)d_out;

  const size_t NX = (size_t)B_ * S_ * H_;  // 8388608 activations
  const size_t NW = (size_t)H_ * H_;       // 1048576 weights

  char* ws = (char*)d_ws;
  const size_t MB = 1024u * 1024u;
  _Float16* x16 = (_Float16*)(ws);            // 16 MB activation staging (reused)
  _Float16* w16 = (_Float16*)(ws + 16 * MB);  //  2 MB weight staging (reused)
  _Float16* Qh  = (_Float16*)(ws + 18 * MB);  // 16 MB [B,NH,S,HD]
  _Float16* Kh  = (_Float16*)(ws + 34 * MB);  // 16 MB [B,NH,S,HD]
  _Float16* Vt  = (_Float16*)(ws + 50 * MB);  // 16 MB [B,NH,HD,S]
  _Float16* ctx = (_Float16*)(ws + 66 * MB);  // 16 MB [B,S,H]  (ends at 82 MB)

  dim3 gGemm(64, 16), bGemm(128);
  dim3 gAttn(S_ / 64, B_ * NH_), bAttn(128);

  auto cvt = [&](const float* src, _Float16* dst, size_t n) {
    cvt16<<<dim3((unsigned)((n / 4 + 255) / 256)), dim3(256), 0, stream>>>(
        src, dst, (int)n);
  };

  cvt(q, x16, NX);
  cvt(Wq, w16, NW);
  gemm_xwt<<<gGemm, bGemm, 0, stream>>>(x16, w16, Qh, nullptr, nullptr, 0);

  cvt(k, x16, NX);
  cvt(Wk, w16, NW);
  gemm_xwt<<<gGemm, bGemm, 0, stream>>>(x16, w16, Kh, nullptr, nullptr, 0);

  cvt(v, x16, NX);
  cvt(Wv, w16, NW);
  gemm_xwt<<<gGemm, bGemm, 0, stream>>>(x16, w16, Vt, nullptr, nullptr, 1);

  attn_fa<<<gAttn, bAttn, 0, stream>>>(Qh, Kh, Vt, mask, ctx);

  cvt(Wo, w16, NW);
  gemm_xwt<<<gGemm, bGemm, 0, stream>>>(ctx, w16, nullptr, out, bo, 2);
}
